// SelfAttention_78563541779079
// MI455X (gfx1250) — matrix operand order repaired and checked by reading, not compiled, not run on hardware
//
#include <hip/hip_runtime.h>
#include <hip/hip_bf16.h>

// ---------------------------------------------------------------------------
// Problem: B=16, C=512, H=W=64 -> N=4096
//   q[b,c]  = wq@y[b] + bq                  (pq is n-invariant!)
//   kk[b,d] = wk@(sum_n x[b]) + N*bk        (energy = q (x) kk, rank-1)
//   attn    = softmax_d(q[c]*kk[d])
//   out     = gamma*((attn@wv)@x + (attn@bv)) + x
// Main cost: one batched bf16-WMMA GEMM  (attn@wv) @ x  (512x512x4096 x 16).
// ---------------------------------------------------------------------------

typedef __attribute__((ext_vector_type(16))) __bf16       v16bf;
typedef __attribute__((ext_vector_type(8)))  float        v8f;
typedef __attribute__((ext_vector_type(4)))  unsigned int v4u;

union FragAB { v4u q[2]; v16bf v; };

static __device__ __forceinline__ unsigned short f2bfu(float x) {
    return __builtin_bit_cast(unsigned short, (__bf16)x);
}

#define CC   512
#define HW   4096
#define NB   16

// ---------------- Kernel 1: row sums of x -> sx[b*C + c] -------------------
__global__ __launch_bounds__(256) void row_sum_kernel(
    const float* __restrict__ x, float* __restrict__ sx)
{
    __shared__ float red[256];
    const long r = blockIdx.x;                 // 0 .. B*C-1
    const float* p = x + r * (long)HW;
    float s = 0.0f;
    for (int i = threadIdx.x * 4; i < HW; i += 1024) {
        float4 v = *(const float4*)(p + i);
        s += v.x + v.y + v.z + v.w;
    }
    red[threadIdx.x] = s;
    __syncthreads();
    for (int off = 128; off > 0; off >>= 1) {
        if ((int)threadIdx.x < off) red[threadIdx.x] += red[threadIdx.x + off];
        __syncthreads();
    }
    if (threadIdx.x == 0) sx[r] = red[0];
}

// ---------------- Kernel 2: q, kk, softmax max & normalizer ----------------
__global__ __launch_bounds__(256) void qk_prep_kernel(
    const float* __restrict__ y,  const float* __restrict__ wq,
    const float* __restrict__ bq, const float* __restrict__ wk,
    const float* __restrict__ bk, const float* __restrict__ sx,
    float* __restrict__ qO, float* __restrict__ kkO,
    float* __restrict__ mxO, float* __restrict__ ZO)
{
    const int b = blockIdx.x, t = threadIdx.x;
    __shared__ float ly[CC], lsx[CC], lkk[CC], lq[CC];
    __shared__ float rmax[256], rmin[256];

    for (int i = t; i < CC; i += 256) { ly[i] = y[b*CC + i]; lsx[i] = sx[b*CC + i]; }
    __syncthreads();

    for (int c = t; c < CC; c += 256) {
        float aq = bq[c];
        float ak = bk[c] * (float)HW;
        const float* wqr = wq + (long)c * CC;
        const float* wkr = wk + (long)c * CC;
        for (int k = 0; k < CC; ++k) {
            aq = fmaf(wqr[k], ly[k],  aq);
            ak = fmaf(wkr[k], lsx[k], ak);
        }
        lq[c] = aq; lkk[c] = ak;
        qO[b*CC + c] = aq; kkO[b*CC + c] = ak;
    }
    __syncthreads();

    float v0 = lkk[t], v1 = lkk[t + 256];
    rmax[t] = fmaxf(v0, v1);
    rmin[t] = fminf(v0, v1);
    __syncthreads();
    for (int off = 128; off > 0; off >>= 1) {
        if (t < off) {
            rmax[t] = fmaxf(rmax[t], rmax[t + off]);
            rmin[t] = fminf(rmin[t], rmin[t + off]);
        }
        __syncthreads();
    }
    const float kmax = rmax[0], kmin = rmin[0];

    for (int c = t; c < CC; c += 256) {
        float qc = lq[c];
        float m  = fmaxf(qc * kmax, qc * kmin);   // handles both signs of qc
        float z  = 0.0f;
        for (int d = 0; d < CC; ++d) z += __expf(qc * lkk[d] - m);
        mxO[b*CC + c] = m;
        ZO[b*CC + c]  = z;
    }
}

// ---------------- Kernel 3: attn rows (bf16) + c0 = attn@bv ----------------
__global__ __launch_bounds__(256) void attn_kernel(
    const float* __restrict__ q,  const float* __restrict__ kk,
    const float* __restrict__ mx, const float* __restrict__ Z,
    const float* __restrict__ bv, __bf16* __restrict__ attn,
    float* __restrict__ c0)
{
    __shared__ float red[256];
    const int bc = blockIdx.x;          // b*C + c
    const int b  = bc >> 9;
    const int t  = threadIdx.x;
    const float qc  = q[bc];
    const float m   = mx[bc];
    const float inv = 1.0f / Z[bc];
    const float* kkb = kk + (b << 9);
    __bf16* arow = attn + (long)bc * CC;

    float acc = 0.0f;
    for (int d = t; d < CC; d += 256) {
        float a = __expf(qc * kkb[d] - m) * inv;
        arow[d] = (__bf16)a;
        acc = fmaf(a, bv[d], acc);
    }
    red[t] = acc;
    __syncthreads();
    for (int off = 128; off > 0; off >>= 1) {
        if (t < off) red[t] += red[t + off];
        __syncthreads();
    }
    if (t == 0) c0[bc] = red[0];
}

// ---------------- Kernel 4: bf16 WMMA GEMM, double-buffered LDS ------------
// C_tile = A(bf16,[m][k]) @ B(f32,[k][n], cvt+transposed into LDS)
//   EPI=false: outH = (bf16)acc                (M = attn @ wv)
//   EPI=true : outF = g*(acc + c0[m]) + x     (final output, fused residual)
template<bool EPI>
__global__ __launch_bounds__(256) void gemm_bf16_kernel(
    const __bf16* __restrict__ A,  long strideAb, int lda,
    const float*  __restrict__ Bg, long strideBb, int ldb,
    float* __restrict__ outF, __bf16* __restrict__ outH,
    long strideOb, int ldo,
    const float* __restrict__ c0, const float* __restrict__ xres,
    const float* __restrict__ gammaPtr)
{
    constexpr int BM = 128, BN = 128, BK = 32, LST = 40;   // 40 bf16 = 80B row
    constexpr int NT = CC / BK;                            // 16 K-steps
    __shared__ __bf16 lsA[2][BM * LST];
    __shared__ __bf16 lsB[2][BN * LST];

    const int bz = blockIdx.z;
    const int m0 = blockIdx.y * BM;
    const int n0 = blockIdx.x * BN;
    const __bf16* Ab = A  + (long)bz * strideAb;
    const float*  Bb = Bg + (long)bz * strideBb;

    const int t    = threadIdx.x;
    const int lane = t & 31, wave = t >> 5;
    const int wm = (wave & 3) * 32;      // 4 waves along M
    const int wn = (wave >> 2) * 64;     // 2 waves along N
    const int l  = lane & 15, hi = lane >> 4;

    const int arow = t >> 1,        aseg = (t & 1) * 16;   // A: 32B per thread
    const int bk2  = (t & 15) * 2,  bnb  = (t >> 4) * 8;   // B: 2 k-rows x 8 n

    v8f acc[2][4];
    #pragma unroll
    for (int i = 0; i < 2; ++i)
        #pragma unroll
        for (int j = 0; j < 4; ++j) acc[i][j] = (v8f)(0.0f);

    // ---- pipeline helpers ----
    auto fetchG = [&](int kt, v4u& a0, v4u& a1,
                      float4& q0, float4& q1, float4& q2, float4& q3) {
        const __bf16* ga = Ab + (long)(m0 + arow) * lda + kt + aseg;
        a0 = *(const v4u*)ga;
        a1 = *(const v4u*)(ga + 8);
        const float* gb = Bb + (long)(kt + bk2) * ldb + (n0 + bnb);
        q0 = *(const float4*)gb;
        q1 = *(const float4*)(gb + 4);
        q2 = *(const float4*)(gb + ldb);
        q3 = *(const float4*)(gb + ldb + 4);
    };
    auto stageT = [&](int buf, v4u a0, v4u a1,
                      float4 q0, float4 q1, float4 q2, float4 q3) {
        __bf16* sa = &lsA[buf][arow * LST + aseg];
        *(v4u*)sa       = a0;
        *(v4u*)(sa + 8) = a1;
        float r0[8] = {q0.x, q0.y, q0.z, q0.w, q1.x, q1.y, q1.z, q1.w};
        float r1[8] = {q2.x, q2.y, q2.z, q2.w, q3.x, q3.y, q3.z, q3.w};
        #pragma unroll
        for (int i = 0; i < 8; ++i) {
            unsigned int p = (unsigned int)f2bfu(r0[i]) |
                             ((unsigned int)f2bfu(r1[i]) << 16);
            *(unsigned int*)(&lsB[buf][(bnb + i) * LST + bk2]) = p;
        }
    };

    // ---- prologue: stage tile 0 ----
    {
        v4u a0, a1; float4 q0, q1, q2, q3;
        fetchG(0, a0, a1, q0, q1, q2, q3);
        stageT(0, a0, a1, q0, q1, q2, q3);
    }
    __syncthreads();

    int cur = 0;
    for (int kt2 = 0; kt2 < NT; ++kt2) {
        const bool more = (kt2 + 1) < NT;

        // 1) issue next tile's global loads ASAP (latency hidden behind WMMA)
        v4u na0, na1; float4 nq0, nq1, nq2, nq3;
        if (more) fetchG((kt2 + 1) * BK, na0, na1, nq0, nq1, nq2, nq3);

        // 2) fragment loads from current buffer
        // A 16x32 frag: lane<16 -> row l, K {0..7,16..23}; lane>=16 -> K {8..15,24..31}
        FragAB fa[2], fb[4];
        #pragma unroll
        for (int i = 0; i < 2; ++i) {
            const __bf16* pa = &lsA[cur][(wm + i*16 + l) * LST + hi * 8];
            fa[i].q[0] = *(const v4u*)pa;
            fa[i].q[1] = *(const v4u*)(pa + 16);
        }
        // B 32x16 frag: lane<16 -> col l, K {0..7,16..23}; lane>=16 -> K {8..15,24..31}
        #pragma unroll
        for (int j = 0; j < 4; ++j) {
            const __bf16* pb = &lsB[cur][(wn + j*16 + l) * LST + hi * 8];
            fb[j].q[0] = *(const v4u*)pb;
            fb[j].q[1] = *(const v4u*)(pb + 16);
        }

        // 3) stage next tile into the other buffer (its last readers finished
        //    before the previous barrier, so no sync needed before writing)
        if (more) stageT(cur ^ 1, na0, na1, nq0, nq1, nq2, nq3);

        // 4) 8 WMMAs on current fragments
        #pragma unroll
        for (int i = 0; i < 2; ++i)
            #pragma unroll
            for (int j = 0; j < 4; ++j)
                acc[i][j] = __builtin_amdgcn_wmma_f32_16x16x32_bf16(
                    false, fa[i].v, false, fb[j].v,
                    (short)0, acc[i][j], false, false);

        // 5) single barrier per K-step
        if (more) __syncthreads();
        cur ^= 1;
    }

    // ---- epilogue: C/D layout -> element (m0+hi*8+r, n0+l) in VGPR r ----
    float g = 0.0f;
    if constexpr (EPI) g = gammaPtr[0];
    #pragma unroll
    for (int i = 0; i < 2; ++i) {
        const int mb = m0 + wm + i*16 + hi*8;
        #pragma unroll
        for (int j = 0; j < 4; ++j) {
            const int nc = n0 + wn + j*16 + l;
            #pragma unroll
            for (int r = 0; r < 8; ++r) {
                const int  m   = mb + r;
                const long idx = (long)bz * strideOb + (long)m * ldo + nc;
                float v = acc[i][j][r];
                if constexpr (EPI) {
                    outF[idx] = g * (v + c0[(bz << 9) + m]) + xres[idx];
                } else {
                    outH[idx] = (__bf16)v;
                }
            }
        }
    }
}

// ---------------------------------------------------------------------------
extern "C" void kernel_launch(void* const* d_in, const int* in_sizes, int n_in,
                              void* d_out, int out_size, void* d_ws, size_t ws_size,
                              hipStream_t stream) {
    (void)in_sizes; (void)n_in; (void)out_size; (void)ws_size;
    const float* x     = (const float*)d_in[0];
    const float* y     = (const float*)d_in[1];
    const float* wq    = (const float*)d_in[2];
    const float* bq    = (const float*)d_in[3];
    const float* wk    = (const float*)d_in[4];
    const float* bk    = (const float*)d_in[5];
    const float* wv    = (const float*)d_in[6];
    const float* bv    = (const float*)d_in[7];
    const float* gamma = (const float*)d_in[8];
    float* out = (float*)d_out;

    // workspace carve (bytes): 6 * B*C f32 scalars, then attn + M in bf16
    char* ws = (char*)d_ws;
    float*  sx   = (float*)(ws);                       // 32 KB
    float*  q    = (float*)(ws + 32768);
    float*  kk   = (float*)(ws + 65536);
    float*  mx   = (float*)(ws + 98304);
    float*  Z    = (float*)(ws + 131072);
    float*  c0   = (float*)(ws + 163840);
    __bf16* attn = (__bf16*)(ws + 196608);             // B*C*C bf16 = 8 MB
    __bf16* Mw   = (__bf16*)(ws + 196608 + (size_t)NB*CC*CC*2);  // 8 MB

    row_sum_kernel<<<NB*CC, 256, 0, stream>>>(x, sx);
    qk_prep_kernel<<<NB, 256, 0, stream>>>(y, wq, bq, wk, bk, sx, q, kk, mx, Z);
    attn_kernel<<<NB*CC, 256, 0, stream>>>(q, kk, mx, Z, bv, attn, c0);

    // M = attn @ wv   (per batch 512x512x512, bf16 out)
    gemm_bf16_kernel<false><<<dim3(CC/128, CC/128, NB), 256, 0, stream>>>(
        attn, (long)CC*CC, CC,
        wv,   0L,          CC,
        nullptr, Mw, (long)CC*CC, CC,
        nullptr, nullptr, nullptr);

    // out = gamma*(M @ x + c0) + x   (per batch 512x512x4096)
    gemm_bf16_kernel<true><<<dim3(HW/128, CC/128, NB), 256, 0, stream>>>(
        Mw, (long)CC*CC, CC,
        x,  (long)CC*HW, HW,
        out, nullptr, (long)CC*HW, HW,
        c0, x, gamma);
}
